// NormalizedCorrelation2D_2439541424700
// MI455X (gfx1250) — matrix operand order repaired and checked by reading, not compiled, not run on hardware
//
#include <hip/hip_runtime.h>
#include <hip/hip_bf16.h>

typedef __bf16 v16bf __attribute__((ext_vector_type(16)));
typedef __bf16 v8bf  __attribute__((ext_vector_type(8)));
typedef float  v8f   __attribute__((ext_vector_type(8)));

constexpr int BATCH = 16;
constexpr int Hh    = 4;
constexpr int Wd    = 900;
constexpr int Cc    = 256;
constexpr int Dk    = Hh * Cc;        // 1024 reduction length
constexpr int NPAD  = Wd / 2;         // 450
constexpr int TI    = 128;            // gram tile rows (i / x1)
constexpr int TJ    = 128;            // gram tile cols (j / x2)
constexpr int KC    = 32;             // K chunk staged in LDS (one wmma K-step)
constexpr int KSTR  = 40;             // padded LDS row stride (bf16 elems), phase-conflict-free b128
constexpr int NCH   = Dk / KC;        // 32 chunks

__global__ void zero_f32_kernel(float* __restrict__ p, int n) {
    int i = blockIdx.x * blockDim.x + threadIdx.x;
    if (i < n) p[i] = 0.0f;
}

// split x -> hi (truncated bf16) + lo (truncated bf16 of remainder);
// Ah*Bh + Ah*Bl + Al*Bh reproduces fp32 products to ~2^-16 relative.
__device__ inline void split4(const float4 v, uint2& hi, uint2& lo) {
    unsigned u0 = __float_as_uint(v.x), u1 = __float_as_uint(v.y);
    unsigned u2 = __float_as_uint(v.z), u3 = __float_as_uint(v.w);
    hi.x = __builtin_amdgcn_perm(u1, u0, 0x07060302u);   // u0.hi16 | u1.hi16<<16
    hi.y = __builtin_amdgcn_perm(u3, u2, 0x07060302u);
    float r0 = v.x - __uint_as_float(u0 & 0xFFFF0000u);
    float r1 = v.y - __uint_as_float(u1 & 0xFFFF0000u);
    float r2 = v.z - __uint_as_float(u2 & 0xFFFF0000u);
    float r3 = v.w - __uint_as_float(u3 & 0xFFFF0000u);
    lo.x = __builtin_amdgcn_perm(__float_as_uint(r1), __float_as_uint(r0), 0x07060302u);
    lo.y = __builtin_amdgcn_perm(__float_as_uint(r3), __float_as_uint(r2), 0x07060302u);
}

__device__ inline v16bf frag_cat(v8bf a, v8bf b) {
    return __builtin_shufflevector(a, b, 0,1,2,3,4,5,6,7,8,9,10,11,12,13,14,15);
}

__device__ inline void stage_global_load(const float* __restrict__ x1,
                                         const float* __restrict__ x2,
                                         size_t bofs, int i0, int j0,
                                         int kc, int lr, int lq, float4 (&st)[8]) {
    const int k0   = kc * KC;
    const int hin  = k0 >> 8;
    const int koff = k0 & 255;
    #pragma unroll
    for (int p = 0; p < 8; ++p) {
        const int row  = p * 32 + lr;              // 0..255: A rows then B rows
        const bool isA = (p < 4);
        const int trow = isA ? row : row - TI;
        const int grow = (isA ? i0 : j0) + trow;
        if (grow < Wd) {
            const float* src = isA ? x1 : x2;
            st[p] = *(const float4*)(src + bofs + (size_t)hin * (Wd * Cc)
                                         + (size_t)grow * Cc + koff + lq * 4);
        } else {
            st[p] = make_float4(0.f, 0.f, 0.f, 0.f);
        }
    }
}

__device__ inline void stage_convert_store(const float4 (&st)[8],
                                           unsigned short* __restrict__ bAh,
                                           unsigned short* __restrict__ bAl,
                                           unsigned short* __restrict__ bBh,
                                           unsigned short* __restrict__ bBl,
                                           int lr, int lq) {
    #pragma unroll
    for (int p = 0; p < 8; ++p) {
        const int row  = p * 32 + lr;
        const bool isA = (p < 4);
        const int trow = isA ? row : row - TI;
        uint2 hi, lo;
        split4(st[p], hi, lo);
        const int off = trow * KSTR + lq * 4;      // 8B aligned
        if (isA) { *(uint2*)(&bAh[off]) = hi; *(uint2*)(&bAl[off]) = lo; }
        else     { *(uint2*)(&bBh[off]) = hi; *(uint2*)(&bBl[off]) = lo; }
    }
}

__global__ __launch_bounds__(256)
void gram_band_bf16_kernel(const float* __restrict__ x1,
                           const float* __restrict__ x2,
                           float* __restrict__ out)
{
    // double-buffered stage: 8 * 10240B = 80KB (+band) -- legal on gfx1250's 320KB LDS
    __shared__ unsigned short sAh[2][TI * KSTR];
    __shared__ unsigned short sAl[2][TI * KSTR];
    __shared__ unsigned short sBh[2][TJ * KSTR];
    __shared__ unsigned short sBl[2][TJ * KSTR];
    __shared__ float band[256];        // diagonals i-j in [-127,127] -> 255 used

    const int ti = blockIdx.x;
    const int tj = blockIdx.y;
    const int b  = blockIdx.z;
    const int i0 = ti * TI;
    const int j0 = tj * TJ;
    const int tid = threadIdx.x;

    band[tid] = 0.0f;

    const int lane = tid & 31;
    const int wv   = tid >> 5;           // wave 0..7 (wave32)
    const int m    = lane & 15;
    const int hh   = lane >> 4;
    const int siG  = wv & 3;             // A row-block: subtiles at siG*32 + {0,16}
    const int sjG  = wv >> 2;            // B col-block: subtiles at sjG*64 + {0,16,32,48}

    v8f c[2][4];
    #pragma unroll
    for (int i2 = 0; i2 < 2; ++i2)
        #pragma unroll
        for (int j4 = 0; j4 < 4; ++j4) c[i2][j4] = (v8f){};

    const size_t bofs = (size_t)b * ((size_t)Hh * Wd * Cc);
    const int lq = tid & 7;              // float4 index within a 32-float row chunk
    const int lr = tid >> 3;             // row 0..31 within a pass

    // prologue: stage chunk 0 into buffer 0
    {
        float4 st[8];
        stage_global_load(x1, x2, bofs, i0, j0, 0, lr, lq, st);
        stage_convert_store(st, sAh[0], sAl[0], sBh[0], sBl[0], lr, lq);
    }
    __syncthreads();

    for (int kc = 0; kc < NCH; ++kc) {
        const int p = kc & 1;

        // (1) issue next chunk's global loads early (latency covered by compute)
        float4 st[8];
        if (kc + 1 < NCH)
            stage_global_load(x1, x2, bofs, i0, j0, kc + 1, lr, lq, st);

        // (2) fragments + 24 wmma from buffer p
        const unsigned short* bAh = sAh[p];
        const unsigned short* bAl = sAl[p];
        const unsigned short* bBh = sBh[p];
        const unsigned short* bBl = sBl[p];

        v16bf Ah[2], Al[2], Bh[4], Bl[4];
        #pragma unroll
        for (int i2 = 0; i2 < 2; ++i2) {
            const int ra = (siG * 32 + i2 * 16 + m) * KSTR;
            // A 16x32 bf16 frag: VGPR0-3 = K 8h..8h+7, VGPR4-7 = K 16+8h..16+8h+7
            Ah[i2] = frag_cat(*(const v8bf*)(&bAh[ra + 8 * hh]),
                              *(const v8bf*)(&bAh[ra + 16 + 8 * hh]));
            Al[i2] = frag_cat(*(const v8bf*)(&bAl[ra + 8 * hh]),
                              *(const v8bf*)(&bAl[ra + 16 + 8 * hh]));
        }
        #pragma unroll
        for (int j4 = 0; j4 < 4; ++j4) {
            const int rb = (sjG * 64 + j4 * 16 + m) * KSTR;
            // B 32x16 bf16 frag: VGPR0-3 = K 8h..8h+7, VGPR4-7 = K 16+8h..16+8h+7
            Bh[j4] = frag_cat(*(const v8bf*)(&bBh[rb + 8 * hh]),
                              *(const v8bf*)(&bBh[rb + 16 + 8 * hh]));
            Bl[j4] = frag_cat(*(const v8bf*)(&bBl[rb + 8 * hh]),
                              *(const v8bf*)(&bBl[rb + 16 + 8 * hh]));
        }
        #pragma unroll
        for (int i2 = 0; i2 < 2; ++i2) {
            #pragma unroll
            for (int j4 = 0; j4 < 4; ++j4) {
                c[i2][j4] = __builtin_amdgcn_wmma_f32_16x16x32_bf16(
                                false, Ah[i2], false, Bh[j4], (short)0, c[i2][j4], false, false);
                c[i2][j4] = __builtin_amdgcn_wmma_f32_16x16x32_bf16(
                                false, Ah[i2], false, Bl[j4], (short)0, c[i2][j4], false, false);
                c[i2][j4] = __builtin_amdgcn_wmma_f32_16x16x32_bf16(
                                false, Al[i2], false, Bh[j4], (short)0, c[i2][j4], false, false);
            }
        }

        // (3) convert + store next chunk into the other buffer; co-executes with XDL drain
        if (kc + 1 < NCH)
            stage_convert_store(st, sAh[p ^ 1], sAl[p ^ 1], sBh[p ^ 1], sBl[p ^ 1], lr, lq);

        __syncthreads();   // write-before-read AND read-before-overwrite for both buffers
    }

    // band reduce: tile element (Mi, Nj) -> diagonal Mi-Nj (+127)
    #pragma unroll
    for (int i2 = 0; i2 < 2; ++i2) {
        #pragma unroll
        for (int j4 = 0; j4 < 4; ++j4) {
            const int Nj = sjG * 64 + j4 * 16 + m;
            #pragma unroll
            for (int r = 0; r < 8; ++r) {
                const int Mi = siG * 32 + i2 * 16 + r + 8 * hh;  // C layout: VGPR r -> M = r + 8h
                unsafeAtomicAdd(&band[Mi - Nj + 127], c[i2][j4][r]);
            }
        }
    }
    __syncthreads();

    if (tid < 255) {
        const int t = i0 - j0 + (tid - 127) - NPAD;  // s = (i - j - 450) mod 900
        const int s = ((t % Wd) + Wd) % Wd;
        unsafeAtomicAdd(&out[b * Wd + s], band[tid]);
    }
}

extern "C" void kernel_launch(void* const* d_in, const int* in_sizes, int n_in,
                              void* d_out, int out_size, void* d_ws, size_t ws_size,
                              hipStream_t stream) {
    const float* x1 = (const float*)d_in[0];
    const float* x2 = (const float*)d_in[1];
    float* out = (float*)d_out;

    const int n = BATCH * Wd; // 14400
    zero_f32_kernel<<<(n + 255) / 256, 256, 0, stream>>>(out, n);

    dim3 grid((Wd + TI - 1) / TI, (Wd + TJ - 1) / TJ, BATCH); // 8 x 8 x 16
    gram_band_bf16_kernel<<<grid, 256, 0, stream>>>(x1, x2, out);
}
